// TrackingLoss_51616916963340
// MI455X (gfx1250) — matrix operand order repaired and checked by reading, not compiled, not run on hardware
//
#include <hip/hip_runtime.h>
#include <math.h>

#define BB 64
#define NN 512
#define DD 512
#define MARGIN 0.3f
#define EPS 1e-6f

typedef __attribute__((ext_vector_type(16))) _Float16 v16h;
typedef __attribute__((ext_vector_type(8)))  float    v8f;

// ---------------------------------------------------------------------------
// Kernel 1: per-row exact fp32 stats: sq[r] = sum(x^2), s[r] = sum(x)
// One wave32 per row, float4 coalesced loads, shuffle reduction.
// ---------------------------------------------------------------------------
__global__ __launch_bounds__(256) void k_row_stats(const float* __restrict__ F,
                                                   float* __restrict__ sq,
                                                   float* __restrict__ ss) {
  const int row  = (blockIdx.x * blockDim.x + threadIdx.x) >> 5;  // 0..B*N-1
  const int lane = threadIdx.x & 31;
  const float4* p = (const float4*)(F + (size_t)row * DD);
  float s = 0.f, q = 0.f;
  #pragma unroll
  for (int j = 0; j < DD / 4 / 32; ++j) {
    float4 v = p[lane + 32 * j];
    s += v.x + v.y + v.z + v.w;
    q = fmaf(v.x, v.x, q); q = fmaf(v.y, v.y, q);
    q = fmaf(v.z, v.z, q); q = fmaf(v.w, v.w, q);
  }
  #pragma unroll
  for (int off = 16; off > 0; off >>= 1) {
    s += __shfl_xor(s, off, 32);
    q += __shfl_xor(q, off, 32);
  }
  if (lane == 0) { ss[row] = s; sq[row] = q; }
}

// ---------------------------------------------------------------------------
// Kernel 2: per-batch mask logic: first positive index, first negative index,
// anchor_ok. One block (256 threads) per batch; ids staged in LDS.
// ---------------------------------------------------------------------------
__global__ __launch_bounds__(256) void k_select(const int* __restrict__ ids,
                                                int* __restrict__ posi,
                                                int* __restrict__ negi,
                                                int* __restrict__ aok) {
  __shared__ int lid[NN];
  const int b = blockIdx.x;
  const int* idb = ids + b * NN;
  lid[threadIdx.x]       = idb[threadIdx.x];
  lid[threadIdx.x + 256] = idb[threadIdx.x + 256];
  __syncthreads();
  int cnt = 0;
  for (int a = 0; a < 2; ++a) {
    const int i   = threadIdx.x + a * 256;
    const int idi = lid[i];
    int fp = -1, fn = -1;
    for (int j = 0; j < NN; ++j) {
      const int idj = lid[j];
      const bool vj = (idj >= 0);
      if (a == 0 && vj) cnt++;
      if (vj && idj == idi && j != i && fp < 0) fp = j;   // pos: same, !self, valid_j
      if (vj && idj != idi && fn < 0) fn = j;             // neg: !same, valid_j
    }
    const bool ok = (idi >= 0) && (fp >= 0) && (fn >= 0) && (cnt >= 2);
    posi[b * NN + i] = (fp < 0) ? 0 : fp;   // argmax of all-false bool == 0
    negi[b * NN + i] = (fn < 0) ? 0 : fn;
    aok [b * NN + i] = ok ? 1 : 0;
  }
}

// extract element i (0..7) of a v8f with a runtime index
__device__ __forceinline__ float v8sel(v8f v, int i) {
  float r = v[0];
  if (i == 1) r = v[1];
  if (i == 2) r = v[2];
  if (i == 3) r = v[3];
  if (i == 4) r = v[4];
  if (i == 5) r = v[5];
  if (i == 6) r = v[6];
  if (i == 7) r = v[7];
  return r;
}

// ---------------------------------------------------------------------------
// Kernel 3: WMMA gathered dot products + hinge.
// One wave32 per 16-anchor tile. A = anchor rows (fp16), B = gathered pos/neg
// rows as columns (fp16), fp32 accumulation, diag(A * B) extracted.
//   d_ap^2 = sq_i + sq_p - 2*dot(i,p) + 2*EPS*(s_i - s_p) + D*EPS^2
// ---------------------------------------------------------------------------
__global__ __launch_bounds__(256) void k_wmma_loss(const float* __restrict__ F,
                                                   const float* __restrict__ sq,
                                                   const float* __restrict__ ss,
                                                   const int* __restrict__ posi,
                                                   const int* __restrict__ negi,
                                                   const int* __restrict__ aok,
                                                   float* __restrict__ contrib) {
  const int wave = (blockIdx.x * blockDim.x + threadIdx.x) >> 5;  // 0..2047
  const int lane = threadIdx.x & 31;
  const int b    = wave >> 5;        // 32 tiles per batch
  const int i0   = (wave & 31) << 4; // anchor tile base
  const int m    = lane & 15;
  const bool hi  = lane >= 16;

  const int arow = i0 + m;
  const int prow = posi[b * NN + arow];
  const int nrow = negi[b * NN + arow];

  // A k order chosen here (contiguous, matches B): lanes 0-15 hold row M=m, K in {0..15};
  // lanes 16-31 hold row M=m, K in {16..31}.
  const float* Abase = F + (size_t)(b * NN + arow) * DD + (hi ? 16 : 0);
  // B k order chosen here (contiguous, matches A): lane = column N=m; lanes 0-15 K=0..15,
  // lanes 16-31 K=16..31, element e <-> K = e + (hi?16:0).
  const float* Pbase = F + (size_t)(b * NN + prow) * DD + (hi ? 16 : 0);
  const float* Nbase = F + (size_t)(b * NN + nrow) * DD + (hi ? 16 : 0);

  v8f cap = {};  // accum for dot(anchor, pos)
  v8f can = {};  // accum for dot(anchor, neg)

  for (int k0 = 0; k0 < DD; k0 += 32) {
    float4 x0 = *(const float4*)(Abase + k0);
    float4 x1 = *(const float4*)(Abase + k0 + 4);
    float4 x2 = *(const float4*)(Abase + k0 + 8);
    float4 x3 = *(const float4*)(Abase + k0 + 12);
    v16h a;
    a[0]=(_Float16)x0.x; a[1]=(_Float16)x0.y; a[2]=(_Float16)x0.z; a[3]=(_Float16)x0.w;
    a[4]=(_Float16)x1.x; a[5]=(_Float16)x1.y; a[6]=(_Float16)x1.z; a[7]=(_Float16)x1.w;
    a[8]=(_Float16)x2.x; a[9]=(_Float16)x2.y; a[10]=(_Float16)x2.z; a[11]=(_Float16)x2.w;
    a[12]=(_Float16)x3.x; a[13]=(_Float16)x3.y; a[14]=(_Float16)x3.z; a[15]=(_Float16)x3.w;

    float4 p0 = *(const float4*)(Pbase + k0);
    float4 p1 = *(const float4*)(Pbase + k0 + 4);
    float4 p2 = *(const float4*)(Pbase + k0 + 8);
    float4 p3 = *(const float4*)(Pbase + k0 + 12);
    v16h bp;
    bp[0]=(_Float16)p0.x; bp[1]=(_Float16)p0.y; bp[2]=(_Float16)p0.z; bp[3]=(_Float16)p0.w;
    bp[4]=(_Float16)p1.x; bp[5]=(_Float16)p1.y; bp[6]=(_Float16)p1.z; bp[7]=(_Float16)p1.w;
    bp[8]=(_Float16)p2.x; bp[9]=(_Float16)p2.y; bp[10]=(_Float16)p2.z; bp[11]=(_Float16)p2.w;
    bp[12]=(_Float16)p3.x; bp[13]=(_Float16)p3.y; bp[14]=(_Float16)p3.z; bp[15]=(_Float16)p3.w;

    float4 n0 = *(const float4*)(Nbase + k0);
    float4 n1 = *(const float4*)(Nbase + k0 + 4);
    float4 n2 = *(const float4*)(Nbase + k0 + 8);
    float4 n3 = *(const float4*)(Nbase + k0 + 12);
    v16h bn;
    bn[0]=(_Float16)n0.x; bn[1]=(_Float16)n0.y; bn[2]=(_Float16)n0.z; bn[3]=(_Float16)n0.w;
    bn[4]=(_Float16)n1.x; bn[5]=(_Float16)n1.y; bn[6]=(_Float16)n1.z; bn[7]=(_Float16)n1.w;
    bn[8]=(_Float16)n2.x; bn[9]=(_Float16)n2.y; bn[10]=(_Float16)n2.z; bn[11]=(_Float16)n2.w;
    bn[12]=(_Float16)n3.x; bn[13]=(_Float16)n3.y; bn[14]=(_Float16)n3.z; bn[15]=(_Float16)n3.w;

    cap = __builtin_amdgcn_wmma_f32_16x16x32_f16(false, a, false, bp,
                                                 (short)0, cap, false, false);
    can = __builtin_amdgcn_wmma_f32_16x16x32_f16(false, a, false, bn,
                                                 (short)0, can, false, false);
  }

  // Diagonal extraction of 16x16 fp32 C/D:
  //  m = 0..7  -> lane m,      vgpr m
  //  m = 8..15 -> lane m+16,   vgpr m-8
  const bool act = (lane < 8) || (lane >= 24);
  if (act) {
    const int md = (lane < 8) ? lane : (lane - 16);  // == m for these lanes
    const int ci = (lane < 8) ? lane : (lane - 24);
    const float dap = v8sel(cap, ci);
    const float dan = v8sel(can, ci);

    const int ia = b * NN + i0 + md;
    const float sqi = sq[ia],           si = ss[ia];
    const float sqp = sq[b * NN + prow], sp = ss[b * NN + prow];
    const float sqn = sq[b * NN + nrow], sn = ss[b * NN + nrow];
    const float cst = (float)DD * EPS * EPS;

    float d2ap = sqi + sqp - 2.f * dap + 2.f * EPS * (si - sp) + cst;
    float d2an = sqi + sqn - 2.f * dan + 2.f * EPS * (si - sn) + cst;
    const float Dap = sqrtf(fmaxf(d2ap, 0.f));
    const float Dan = sqrtf(fmaxf(d2an, 0.f));
    const float per = fmaxf(Dap - Dan + MARGIN, 0.f);
    contrib[ia] = aok[ia] ? per : 0.f;
  }
}

// ---------------------------------------------------------------------------
// Kernel 4: deterministic final reduction -> (loss, loss, 0)
// ---------------------------------------------------------------------------
__global__ __launch_bounds__(256) void k_finalize(const float* __restrict__ contrib,
                                                  const int* __restrict__ aok,
                                                  float* __restrict__ out) {
  __shared__ float sf[256];
  __shared__ int   sc[256];
  float t = 0.f; int c = 0;
  for (int i = threadIdx.x; i < BB * NN; i += 256) {
    t += contrib[i];
    c += aok[i];
  }
  sf[threadIdx.x] = t; sc[threadIdx.x] = c;
  __syncthreads();
  for (int off = 128; off > 0; off >>= 1) {
    if (threadIdx.x < off) {
      sf[threadIdx.x] += sf[threadIdx.x + off];
      sc[threadIdx.x] += sc[threadIdx.x + off];
    }
    __syncthreads();
  }
  if (threadIdx.x == 0) {
    const int n = sc[0];
    const float loss = (n > 0) ? (sf[0] / (float)n) : 0.f;
    out[0] = loss;  // tracking_loss
    out[1] = loss;  // loss_triplet
    out[2] = 0.f;   // loss_id
  }
}

extern "C" void kernel_launch(void* const* d_in, const int* in_sizes, int n_in,
                              void* d_out, int out_size, void* d_ws, size_t ws_size,
                              hipStream_t stream) {
  const float* F   = (const float*)d_in[0];   // (64, 512, 512) fp32
  const int*   ids = (const int*)d_in[1];     // (64, 512) int32
  float* out = (float*)d_out;                 // 3 fp32 scalars

  // workspace layout (all 4-byte elements), total 768 KiB
  float* sq      = (float*)d_ws;                       // B*N
  float* ss      = sq + BB * NN;                       // B*N
  int*   posi    = (int*)(ss + BB * NN);               // B*N
  int*   negi    = posi + BB * NN;                     // B*N
  int*   aok     = negi + BB * NN;                     // B*N
  float* contrib = (float*)(aok + BB * NN);            // B*N

  // 1) exact fp32 row stats: 32768 rows, 1 wave each, 8 waves/block
  k_row_stats<<<(BB * NN) / 8, 256, 0, stream>>>(F, sq, ss);

  // 2) mask / first-index selection: 1 block per batch
  k_select<<<BB, 256, 0, stream>>>(ids, posi, negi, aok);

  // 3) WMMA gathered dots + hinge: 2048 tiles * 32 lanes = 65536 threads
  k_wmma_loss<<<256, 256, 0, stream>>>(F, sq, ss, posi, negi, aok, contrib);

  // 4) deterministic scalar reduction
  k_finalize<<<1, 256, 0, stream>>>(contrib, aok, out);
}
